// BotRGCN_40467181863061
// MI455X (gfx1250) — matrix operand order repaired and checked by reading, not compiled, not run on hardware
//
#include <hip/hip_runtime.h>
#include <hip/hip_bf16.h>

typedef __bf16 bf16_t;
typedef __attribute__((ext_vector_type(16))) __bf16 v16bf;
typedef __attribute__((ext_vector_type(8)))  float  v8f;

#define NSLOPE 0.01f

__device__ __forceinline__ float lrelu_f(float v) { return v >= 0.0f ? v : NSLOPE * v; }

__device__ __forceinline__ void atomAddF(float* p, float v) {
  __hip_atomic_fetch_add(p, v, __ATOMIC_RELAXED, __HIP_MEMORY_SCOPE_AGENT);
}

// ---------------------------------------------------------------------------
// WMMA tile helpers for v_wmma_f32_16x16x32_bf16 (layouts per CDNA5 ISA 7.12.2)
// ---------------------------------------------------------------------------

// A: 16(M) x 32(K) bf16 from f32 rows with per-row scale; b64 loads + pk-cvt.
// lanes 0-15: M=lane&15, VGPR i -> K = 16*(i>>2) + 8*(lane>=16) + 2*(i&3).
__device__ __forceinline__ v16bf load_A_tile(const float* __restrict__ A, int lda,
                                             int m0, int k0, int M, float scale) {
  const int lane = threadIdx.x & 31;
  int m = m0 + (lane & 15);
  if (m >= M) m = M - 1;  // clamp; stores are guarded
  const int hi = lane >> 4;
  const float* row = A + (size_t)m * lda;
#pragma clang diagnostic push
#pragma clang diagnostic ignored "-Wuninitialized"
  v16bf a;
#pragma unroll
  for (int i = 0; i < 8; ++i) {
    const int k = k0 + ((i >> 2) << 4) + (hi << 3) + ((i & 3) << 1);
    const float2 p = *(const float2*)(row + k);  // k even -> 8B aligned
    a[2 * i]     = (bf16_t)(p.x * scale);
    a[2 * i + 1] = (bf16_t)(p.y * scale);
  }
#pragma clang diagnostic pop
  return a;
}

// B: pre-packed register-layout tiles: [kt][nt][lane][16 bf16]; one 32B load.
__device__ __forceinline__ v16bf load_B_packed(const bf16_t* __restrict__ Wt,
                                               int kt, int ntiles, int t) {
  const int lane = threadIdx.x & 31;
  const size_t off = ((((size_t)kt * ntiles + t) * 32) + lane) * 16;
  return *(const v16bf*)(Wt + off);
}

// Store NT accumulated 16x16 tiles; fast unguarded path for full tiles.
template <int NT>
__device__ __forceinline__ void store_tiles(float* __restrict__ C, int ldc, int col0,
                                            int m0, int M, const v8f* c,
                                            const float* __restrict__ bias, int relu) {
  const int lane = threadIdx.x & 31;
  const int nl = lane & 15;
  const int hi = lane >> 4;
  if (m0 + 16 <= M) {
#pragma unroll
    for (int t = 0; t < NT; ++t) {
      const float bv = bias[t * 16 + nl];
#pragma unroll
      for (int r = 0; r < 8; ++r) {
        float v = c[t][r] + bv;
        if (relu) v = lrelu_f(v);
        C[(size_t)(m0 + r + (hi << 3)) * ldc + col0 + t * 16 + nl] = v;
      }
    }
  } else {
#pragma unroll
    for (int t = 0; t < NT; ++t) {
      const float bv = bias[t * 16 + nl];
#pragma unroll
      for (int r = 0; r < 8; ++r) {
        const int m = m0 + r + (hi << 3);
        if (m < M) {
          float v = c[t][r] + bv;
          if (relu) v = lrelu_f(v);
          C[(size_t)m * ldc + col0 + t * 16 + nl] = v;
        }
      }
    }
  }
}

// ---------------------------------------------------------------------------
// Kernels
// ---------------------------------------------------------------------------

// Pack f32 row-major W[K x Nw] into WMMA B-register layout (bf16), one dword/thread.
__global__ void k_pack_B(const float* __restrict__ W, int K, int Nw,
                         bf16_t* __restrict__ out) {
  const int gid = blockIdx.x * blockDim.x + threadIdx.x;
  const int total = (K * Nw) >> 1;
  if (gid >= total) return;
  const int i = gid & 7;
  const int lane = (gid >> 3) & 31;
  const int tile = gid >> 8;
  const int ntiles = Nw >> 4;
  const int nt = tile % ntiles;
  const int kt = tile / ntiles;
  const int n = nt * 16 + (lane & 15);
  const int k = kt * 32 + ((i >> 2) << 4) + ((lane >> 4) << 3) + ((i & 3) << 1);
  out[(size_t)gid * 2]     = (bf16_t)W[(size_t)k * Nw + n];
  out[(size_t)gid * 2 + 1] = (bf16_t)W[(size_t)(k + 1) * Nw + n];
}

// GEMM: C[:, col0:col0+16*NT] = act(A[MxK] @ W + bias); each wave owns 16 rows x all cols.
template <int NT>
__global__ void __launch_bounds__(256) k_gemm_bf16(
    const float* __restrict__ A, int M, int K,
    const bf16_t* __restrict__ Wt,
    const float* __restrict__ bias,
    float* __restrict__ C, int ldc, int col0, int relu) {
  const int wave = threadIdx.x >> 5;
  const int m0 = (blockIdx.x * 8 + wave) * 16;
  v8f c[NT];
#pragma unroll
  for (int t = 0; t < NT; ++t) c[t] = (v8f){};
  for (int k0 = 0; k0 < K; k0 += 32) {
    const v16bf a = load_A_tile(A, K, m0, k0, M, 1.0f);
    const int kt = k0 >> 5;
#pragma unroll
    for (int t = 0; t < NT; ++t) {
      const v16bf b = load_B_packed(Wt, kt, NT, t);
      c[t] = __builtin_amdgcn_wmma_f32_16x16x32_bf16(false, a, false, b, (short)0, c[t],
                                                     false, false);
    }
  }
  store_tiles<NT>(C, ldc, col0, m0, M, c, bias, relu);
}

// Small encoder branches: num_prop (K=5) -> cols 64..95, cat_prop (K=3) -> cols 96..127
__global__ void k_enc_small(const float* __restrict__ nump, const float* __restrict__ catp,
                            const float* __restrict__ Wn, const float* __restrict__ bn,
                            const float* __restrict__ Wc, const float* __restrict__ bc,
                            float* __restrict__ H, int M) {
  int idx = blockIdx.x * blockDim.x + threadIdx.x;
  int row = idx >> 6;
  int c = idx & 63;
  if (row >= M) return;
  float v;
  int outc;
  if (c < 32) {
    v = bn[c];
#pragma unroll
    for (int k = 0; k < 5; ++k) v += nump[(size_t)row * 5 + k] * Wn[k * 32 + c];
    outc = 64 + c;
  } else {
    c -= 32;
    v = bc[c];
#pragma unroll
    for (int k = 0; k < 3; ++k) v += catp[(size_t)row * 3 + k] * Wc[k * 32 + c];
    outc = 96 + c;
  }
  H[(size_t)row * 128 + outc] = lrelu_f(v);
}

// Per-(dst, relation) in-degree counts
__global__ void k_count(const int* __restrict__ dst, const int* __restrict__ et,
                        float* __restrict__ cnt, int E, int M) {
  int e = blockIdx.x * blockDim.x + threadIdx.x;
  if (e >= E) return;
  atomAddF(&cnt[(size_t)et[e] * M + dst[e]], 1.0f);
}

__global__ void k_inv(const float* __restrict__ cnt, float* __restrict__ inv, int n) {
  int i = blockIdx.x * blockDim.x + threadIdx.x;
  if (i < n) inv[i] = 1.0f / fmaxf(cnt[i], 1.0f);
}

// Edge scatter: one wave per edge, 128 dims = 4 per lane; agg layout [2][M][128]
__global__ void __launch_bounds__(256) k_scatter(
    const float* __restrict__ X, const int* __restrict__ src, const int* __restrict__ dst,
    const int* __restrict__ et, float* __restrict__ agg, int E, int M) {
  int gid = blockIdx.x * blockDim.x + threadIdx.x;
  int e = gid >> 5;
  int lane = gid & 31;
  if (e >= E) return;
  int s = src[e], d = dst[e], r = et[e];
  const float* xs = X + (size_t)s * 128;
  float* ag = agg + ((size_t)r * M + d) * 128;
#pragma unroll
  for (int j = 0; j < 4; ++j) {
    int dim = lane + j * 32;
    atomAddF(&ag[dim], xs[dim]);
  }
}

// Fused RGCN combine: Y = X@root + (agg0*inv0)@W0 + (agg1*inv1)@W1 + bias.
// Each wave: 16 rows x 128 cols; A tiles loaded once per k-step, reused over 8 N tiles.
__global__ void __launch_bounds__(256) k_rgcn_combine(
    const float* __restrict__ X, const float* __restrict__ agg,
    const float* __restrict__ inv,
    const bf16_t* __restrict__ roott, const bf16_t* __restrict__ W0t,
    const bf16_t* __restrict__ W1t, const float* __restrict__ bias,
    float* __restrict__ Y, int M) {
  const int wave = threadIdx.x >> 5;
  const int m0 = (blockIdx.x * 8 + wave) * 16;
  const int lane = threadIdx.x & 31;
  int mr = m0 + (lane & 15);
  if (mr >= M) mr = M - 1;
  const float s0 = inv[mr];
  const float s1 = inv[(size_t)M + mr];
  const float* agg0 = agg;
  const float* agg1 = agg + (size_t)M * 128;

  v8f c[8];
#pragma unroll
  for (int t = 0; t < 8; ++t) c[t] = (v8f){};

#pragma unroll
  for (int k0 = 0; k0 < 128; k0 += 32) {
    const int kt = k0 >> 5;
    const v16bf ax = load_A_tile(X, 128, m0, k0, M, 1.0f);
    const v16bf a0 = load_A_tile(agg0, 128, m0, k0, M, s0);
    const v16bf a1 = load_A_tile(agg1, 128, m0, k0, M, s1);
#pragma unroll
    for (int t = 0; t < 8; ++t) {
      v16bf b = load_B_packed(roott, kt, 8, t);
      c[t] = __builtin_amdgcn_wmma_f32_16x16x32_bf16(false, ax, false, b, (short)0, c[t],
                                                     false, false);
      b = load_B_packed(W0t, kt, 8, t);
      c[t] = __builtin_amdgcn_wmma_f32_16x16x32_bf16(false, a0, false, b, (short)0, c[t],
                                                     false, false);
      b = load_B_packed(W1t, kt, 8, t);
      c[t] = __builtin_amdgcn_wmma_f32_16x16x32_bf16(false, a1, false, b, (short)0, c[t],
                                                     false, false);
    }
  }
  store_tiles<8>(Y, 128, 0, m0, M, c, bias, 0);
}

// Output head: out = lrelu(X @ W1[128x64] + b1) @ W2[64x2] + b2; one wave per row.
__global__ void __launch_bounds__(128) k_out_head(
    const float* __restrict__ X, const float* __restrict__ W1, const float* __restrict__ b1,
    const float* __restrict__ W2, const float* __restrict__ b2,
    float* __restrict__ out, int M) {
  const int wave = threadIdx.x >> 5;
  const int lane = threadIdx.x & 31;
  const int row = blockIdx.x * 4 + wave;
  if (row >= M) return;
  const float* x = X + (size_t)row * 128;
  float h0 = b1[lane], h1 = b1[lane + 32];
  for (int k = 0; k < 128; ++k) {
    const float xv = x[k];
    h0 = fmaf(xv, W1[k * 64 + lane], h0);
    h1 = fmaf(xv, W1[k * 64 + lane + 32], h1);
  }
  h0 = lrelu_f(h0);
  h1 = lrelu_f(h1);
  float p0 = h0 * W2[lane * 2 + 0] + h1 * W2[(lane + 32) * 2 + 0];
  float p1 = h0 * W2[lane * 2 + 1] + h1 * W2[(lane + 32) * 2 + 1];
#pragma unroll
  for (int off = 16; off > 0; off >>= 1) {
    p0 += __shfl_down(p0, off, 32);
    p1 += __shfl_down(p1, off, 32);
  }
  if (lane == 0) {
    out[(size_t)row * 2 + 0] = p0 + b2[0];
    out[(size_t)row * 2 + 1] = p1 + b2[1];
  }
}

// ---------------------------------------------------------------------------
// Host launcher
// ---------------------------------------------------------------------------
extern "C" void kernel_launch(void* const* d_in, const int* in_sizes, int n_in,
                              void* d_out, int out_size, void* d_ws, size_t ws_size,
                              hipStream_t stream) {
  const float* des   = (const float*)d_in[0];
  const float* tweet = (const float*)d_in[1];
  const float* nump  = (const float*)d_in[2];
  const float* catp  = (const float*)d_in[3];
  const int*   ei    = (const int*)d_in[4];
  const int*   et    = (const int*)d_in[5];
  const float* W_des = (const float*)d_in[6];
  const float* b_des = (const float*)d_in[7];
  const float* W_tw  = (const float*)d_in[8];
  const float* b_tw  = (const float*)d_in[9];
  const float* W_num = (const float*)d_in[10];
  const float* b_num = (const float*)d_in[11];
  const float* W_cat = (const float*)d_in[12];
  const float* b_cat = (const float*)d_in[13];
  const float* W_in  = (const float*)d_in[14];
  const float* b_in  = (const float*)d_in[15];
  const float* rgcnW = (const float*)d_in[16];  // [2,128,128]
  const float* root  = (const float*)d_in[17];
  const float* rbias = (const float*)d_in[18];
  const float* W_o1  = (const float*)d_in[19];
  const float* b_o1  = (const float*)d_in[20];
  const float* W_o2  = (const float*)d_in[21];
  const float* b_o2  = (const float*)d_in[22];

  const int M = in_sizes[0] / 768;  // N nodes
  const int E = in_sizes[5];        // edges
  const int* src = ei;
  const int* dst = ei + E;

  // ---- workspace carve-out ----
  char* ws = (char*)d_ws;
  size_t off = 0;
  auto take = [&](size_t bytes) -> char* {
    char* p = ws + off;
    off += (bytes + 255) & ~(size_t)255;
    return p;
  };
  float*  H     = (float*)take((size_t)M * 128 * 4);
  float*  X     = (float*)take((size_t)M * 128 * 4);
  float*  agg   = (float*)take((size_t)2 * M * 128 * 4);
  float*  cnt   = (float*)take((size_t)2 * M * 4);
  float*  inv   = (float*)take((size_t)2 * M * 4);
  bf16_t* Wdesb = (bf16_t*)take((size_t)768 * 32 * 2);
  bf16_t* Wtwb  = (bf16_t*)take((size_t)768 * 32 * 2);
  bf16_t* Winb  = (bf16_t*)take((size_t)128 * 128 * 2);
  bf16_t* rootb = (bf16_t*)take((size_t)128 * 128 * 2);
  bf16_t* W0b   = (bf16_t*)take((size_t)128 * 128 * 2);
  bf16_t* W1b   = (bf16_t*)take((size_t)128 * 128 * 2);

  // ---- weight packing into WMMA B-register layout ----
  auto pack = [&](const float* w, int K, int Nw, bf16_t* o) {
    int tot = (K * Nw) >> 1;
    k_pack_B<<<(tot + 255) / 256, 256, 0, stream>>>(w, K, Nw, o);
  };
  pack(W_des, 768, 32, Wdesb);
  pack(W_tw, 768, 32, Wtwb);
  pack(W_in, 128, 128, Winb);
  pack(root, 128, 128, rootb);
  pack(rgcnW, 128, 128, W0b);
  pack(rgcnW + 128 * 128, 128, 128, W1b);

  const int mblocks = (M + 127) / 128;  // 8 waves x 16 rows per block

  // ---- encoder: H = concat(lrelu branches) ----
  k_enc_small<<<((size_t)M * 64 + 255) / 256, 256, 0, stream>>>(nump, catp, W_num, b_num,
                                                                W_cat, b_cat, H, M);
  k_gemm_bf16<2><<<mblocks, 256, 0, stream>>>(des, M, 768, Wdesb, b_des, H, 128, 0, 1);
  k_gemm_bf16<2><<<mblocks, 256, 0, stream>>>(tweet, M, 768, Wtwb, b_tw, H, 128, 32, 1);
  // X = lrelu(H @ W_in + b_in)
  k_gemm_bf16<8><<<mblocks, 256, 0, stream>>>(H, M, 128, Winb, b_in, X, 128, 0, 1);

  // ---- graph degree counts (static across layers) ----
  hipMemsetAsync(cnt, 0, (size_t)2 * M * 4, stream);
  k_count<<<(E + 255) / 256, 256, 0, stream>>>(dst, et, cnt, E, M);
  k_inv<<<(2 * M + 255) / 256, 256, 0, stream>>>(cnt, inv, 2 * M);

  // ---- RGCN layer 1: X -> H ----
  hipMemsetAsync(agg, 0, (size_t)2 * M * 128 * 4, stream);
  k_scatter<<<((size_t)E * 32 + 255) / 256, 256, 0, stream>>>(X, src, dst, et, agg, E, M);
  k_rgcn_combine<<<mblocks, 256, 0, stream>>>(X, agg, inv, rootb, W0b, W1b, rbias, H, M);

  // ---- RGCN layer 2: H -> X ----
  hipMemsetAsync(agg, 0, (size_t)2 * M * 128 * 4, stream);
  k_scatter<<<((size_t)E * 32 + 255) / 256, 256, 0, stream>>>(H, src, dst, et, agg, E, M);
  k_rgcn_combine<<<mblocks, 256, 0, stream>>>(H, agg, inv, rootb, W0b, W1b, rbias, X, M);

  // ---- output head ----
  k_out_head<<<(M + 3) / 4, 128, 0, stream>>>(X, W_o1, b_o1, W_o2, b_o2, (float*)d_out, M);
}
